// ScaledDotProductAttention_70755291234465
// MI455X (gfx1250) — compile-verified, matrix operand order checked by reading, not run on hardware
//
#include <hip/hip_runtime.h>
#include <cstddef>
#include <cstdint>

// ---------------------------------------------------------------------------
// Scaled dot-product attention forward (context + full attn matrix), MI455X.
// B=2 H=16 S=2048 D=64, fp32 in/out.  f16 WMMA (f32 accum) for both matmuls.
// Roofline: ~700 MB HBM traffic (512 MB attn stream dominates) => NT stores;
// compute ~34 GFLOP via v_wmma_f32_16x16x32_f16 (8x denser than f32 WMMA).
// ---------------------------------------------------------------------------

typedef _Float16 v16h __attribute__((ext_vector_type(16)));
typedef float    v8f  __attribute__((ext_vector_type(8)));

constexpr int Bc = 2, Hc = 16, Sc = 2048, Dc = 64;
constexpr int BH     = Bc * Hc;          // 32
constexpr int QTILES = Sc / 16;          // 128 query tiles of 16 rows
constexpr int THREADS = 128;             // 4 waves (wave32)
constexpr long long CTX_TOTAL = (long long)BH * Sc * Dc;       // 4,194,304
constexpr int SMEM_FLOATS = 16 * Sc + 16;                      // scores + inv
constexpr int SMEM_BYTES  = SMEM_FLOATS * (int)sizeof(float);  // ~128.06 KB

// f16 A/B operand layout for V_WMMA_F32_16X16X32_F16 (wave32):
//   lane = (row|col)&15, hi = lane>>4 selects K-half.
//   element e in v16h:  K = (e&7) + 8*hi + 16*(e>>3)
// Pairs (2j,2j+1) are K-contiguous -> float2 loads where source is contiguous.
__device__ __forceinline__ v16h load_rowmaj_f16(const float* p, int hi) {
  v16h r;
#pragma unroll
  for (int j = 0; j < 8; ++j) {
    int k = 2 * j + 8 * hi + ((j & 4) ? 8 : 0);
    float2 f = *reinterpret_cast<const float2*>(p + k);
    r[2 * j]     = (_Float16)f.x;
    r[2 * j + 1] = (_Float16)f.y;
  }
  return r;
}

// Column-major source (stride between consecutive K elements), e.g. V rows.
__device__ __forceinline__ v16h load_colmaj_f16(const float* p, int hi, int strideElems) {
  v16h r;
#pragma unroll
  for (int e = 0; e < 16; ++e) {
    int k = (e & 7) + 8 * hi + 16 * (e >> 3);
    r[e] = (_Float16)p[(size_t)k * strideElems];
  }
  return r;
}

__global__ __launch_bounds__(THREADS, 1)
void ScaledDotProductAttention_kernel(const float* __restrict__ Q,
                                      const float* __restrict__ K,
                                      const float* __restrict__ V,
                                      const unsigned char* __restrict__ Msk,
                                      float* __restrict__ out) {
  extern __shared__ float smem[];
  float* sc  = smem;              // [16][2048] score / prob tile, reused for partials
  float* inv = smem + 16 * Sc;    // [16] per-row 1/sum

  const int tid  = threadIdx.x;
  const int wid  = tid >> 5;      // wave id 0..3
  const int lane = tid & 31;
  const int hi   = lane >> 4;     // K-half select
  const int col  = lane & 15;     // N / row-within-tile

  const int bh = blockIdx.x / QTILES;
  const int qt = blockIdx.x % QTILES;
  const int q0 = qt * 16;

  // ---- Q tile as WMMA A operands (16x64 -> two 16x32 f16 chunks) -----------
  const float* qp = Q + ((size_t)bh * Sc + q0 + col) * Dc;
  const v16h qa0 = load_rowmaj_f16(qp,      hi);
  const v16h qa1 = load_rowmaj_f16(qp + 32, hi);

  // ---- Phase 1: S = (Q K^T) * scale, masked; waves stripe key tiles --------
  for (int kt = wid; kt < QTILES; kt += 4) {
    const float* kp = K + ((size_t)bh * Sc + kt * 16 + col) * Dc;
    if (kt + 4 < QTILES)                      // prefetch next K tile (L2 hint)
      __builtin_prefetch(kp + 4 * 16 * Dc, 0, 3);
    v16h kb0 = load_rowmaj_f16(kp,      hi);
    v16h kb1 = load_rowmaj_f16(kp + 32, hi);
    v8f acc = (v8f){0.f, 0.f, 0.f, 0.f, 0.f, 0.f, 0.f, 0.f};
    acc = __builtin_amdgcn_wmma_f32_16x16x32_f16(false, qa0, false, kb0,
                                                 (short)0, acc, false, false);
    acc = __builtin_amdgcn_wmma_f32_16x16x32_f16(false, qa1, false, kb1,
                                                 (short)0, acc, false, false);
#pragma unroll
    for (int r = 0; r < 8; ++r) {
      const int m = r + 8 * hi;          // C/D layout: VGPR r, lane half
      const int n = kt * 16 + col;
      float s = acc[r] * 0.125f;         // 1/sqrt(64)
      // mask is a 128 MB single-use stream -> non-temporal load
      if (__builtin_nontemporal_load(&Msk[((size_t)bh * Sc + q0 + m) * Sc + n]))
        s = -1.0e9f;
      sc[m * Sc + n] = s;
    }
  }
  __syncthreads();

  // ---- Phase 2: row softmax statistics (8 threads per row) -----------------
  {
    const int row = tid >> 3, sub = tid & 7;
    float mx = -3.0e38f;
    for (int k2 = sub; k2 < Sc; k2 += 8) mx = fmaxf(mx, sc[row * Sc + k2]);
#pragma unroll
    for (int o = 1; o < 8; o <<= 1) mx = fmaxf(mx, __shfl_xor(mx, o, 32));
    float sum = 0.f;
    for (int k2 = sub; k2 < Sc; k2 += 8) {
      float e = __expf(sc[row * Sc + k2] - mx);
      sc[row * Sc + k2] = e;
      sum += e;
    }
#pragma unroll
    for (int o = 1; o < 8; o <<= 1) sum += __shfl_xor(sum, o, 32);
    if (sub == 0) inv[row] = 1.0f / sum;
  }
  __syncthreads();

  // ---- Phase 3: normalize; stream attn to HBM non-temporal (512 MB >> L2) --
  {
    float* attn = out + CTX_TOTAL + ((size_t)bh * Sc + q0) * Sc;
    for (int idx = tid; idx < 16 * Sc; idx += THREADS) {
      float p = sc[idx] * inv[idx >> 11];
      sc[idx] = p;
      __builtin_nontemporal_store(p, &attn[idx]);  // th:NT, coalesced
    }
  }
  __syncthreads();

  // ---- Phase 4: context partial = P(f16) @ V(f16), 32-key chunks per wave --
  v8f accd[4];
#pragma unroll
  for (int nt = 0; nt < 4; ++nt)
    accd[nt] = (v8f){0.f, 0.f, 0.f, 0.f, 0.f, 0.f, 0.f, 0.f};

  for (int c = 0; c < 16; ++c) {
    const int kc = wid * 16 + c;                         // key chunk (32 keys)
    v16h a = load_rowmaj_f16(sc + col * Sc + kc * 32, hi);
    const float* vbase = V + ((size_t)bh * Sc + kc * 32) * Dc;
    if (c + 1 < 16)                         // prefetch next V chunk (32x64 f32)
      __builtin_prefetch(vbase + 32 * Dc + lane * 64, 0, 3);
#pragma unroll
    for (int nt = 0; nt < 4; ++nt) {
      v16h b = load_colmaj_f16(vbase + nt * 16 + col, hi, Dc);
      accd[nt] = __builtin_amdgcn_wmma_f32_16x16x32_f16(false, a, false, b,
                                                        (short)0, accd[nt],
                                                        false, false);
    }
  }
  __syncthreads();                 // all waves done reading P from LDS

  // stash per-wave partials in (now free) score buffer: [wave][16][64]
#pragma unroll
  for (int nt = 0; nt < 4; ++nt)
#pragma unroll
    for (int r = 0; r < 8; ++r) {
      const int m = r + 8 * hi;
      sc[(wid * 16 + m) * Dc + nt * 16 + col] = accd[nt][r];
    }
  __syncthreads();

  // ---- Phase 5: cross-wave reduce, coalesced context store -----------------
  {
    float* ctx = out + ((size_t)bh * Sc + q0) * Dc;
    for (int idx = tid; idx < 16 * Dc; idx += THREADS) {
      ctx[idx] = sc[idx] + sc[1024 + idx] + sc[2048 + idx] + sc[3072 + idx];
    }
  }
}

// ---------------------------------------------------------------------------
extern "C" void kernel_launch(void* const* d_in, const int* in_sizes, int n_in,
                              void* d_out, int out_size, void* d_ws, size_t ws_size,
                              hipStream_t stream) {
  (void)in_sizes; (void)n_in; (void)out_size; (void)d_ws; (void)ws_size;
  const float*         Q   = (const float*)d_in[0];
  const float*         K   = (const float*)d_in[1];
  const float*         V   = (const float*)d_in[2];
  const unsigned char* Msk = (const unsigned char*)d_in[3];   // jnp.bool_ -> 1B
  float* out = (float*)d_out;

  // 128 KB dynamic LDS (WGP supports up to 320 KB) — opt in above default cap.
  hipFuncSetAttribute(reinterpret_cast<const void*>(ScaledDotProductAttention_kernel),
                      hipFuncAttributeMaxDynamicSharedMemorySize, SMEM_BYTES);

  dim3 grid(BH * QTILES);   // 4096 workgroups
  ScaledDotProductAttention_kernel<<<grid, THREADS, SMEM_BYTES, stream>>>(
      Q, K, V, Msk, out);
}
